// HarmonicFrequencyAttention_29618094473990
// MI455X (gfx1250) — matrix operand order repaired and checked by reading, not compiled, not run on hardware
//
#include <hip/hip_runtime.h>
#include <hip/hip_bf16.h>

// HarmonicFrequencyAttention for MI455X (gfx1250, wave32, WMMA).
// N=B*T=256 sequences, F=512 freq bins, D=256, H=4 heads, HD=64.
// bf16 WMMA everywhere (f32 accum), fused QKV projection (x read once),
// flash-style online softmax, K/V staged into LDS with async-to-LDS DMA.

typedef __attribute__((ext_vector_type(16))) __bf16 v16bf;
typedef __attribute__((ext_vector_type(8)))  float  v8f;

#define NSEQ 256
#define FB   512
#define DM   256
#define NH_  4
#define HD_  64
#define NEGC (-1.0e9f)

union BfPair { unsigned int u; __bf16 h[2]; };

__device__ __forceinline__ v8f wmma_bf16(v16bf a, v16bf b, v8f c) {
  // D = A(16x32 bf16) x B(32x16 bf16) + C(16x16 f32)
  return __builtin_amdgcn_wmma_f32_16x16x32_bf16(
      /*neg_a=*/false, a, /*neg_b=*/false, b,
      /*c_mod=*/(short)0, c, /*reuse_a=*/false, /*reuse_b=*/false);
}

// Butterfly reductions over the 16-lane half-groups (rows of a WMMA C tile
// live within one half). ds_swizzle group-of-32 xor patterns; xor<16 stays
// inside each half; 0x401F (xor 16) merges halves for the wave-wide OR.
__device__ __forceinline__ float rowmax16(float x) {
  x = fmaxf(x, __int_as_float(__builtin_amdgcn_ds_swizzle(__float_as_int(x), 0x041F)));
  x = fmaxf(x, __int_as_float(__builtin_amdgcn_ds_swizzle(__float_as_int(x), 0x081F)));
  x = fmaxf(x, __int_as_float(__builtin_amdgcn_ds_swizzle(__float_as_int(x), 0x101F)));
  x = fmaxf(x, __int_as_float(__builtin_amdgcn_ds_swizzle(__float_as_int(x), 0x201F)));
  return x;
}
__device__ __forceinline__ float rowsum16(float x) {
  x += __int_as_float(__builtin_amdgcn_ds_swizzle(__float_as_int(x), 0x041F));
  x += __int_as_float(__builtin_amdgcn_ds_swizzle(__float_as_int(x), 0x081F));
  x += __int_as_float(__builtin_amdgcn_ds_swizzle(__float_as_int(x), 0x101F));
  x += __int_as_float(__builtin_amdgcn_ds_swizzle(__float_as_int(x), 0x201F));
  return x;
}
__device__ __forceinline__ int wave_or32(int x) {
  x |= __builtin_amdgcn_ds_swizzle(x, 0x041F);
  x |= __builtin_amdgcn_ds_swizzle(x, 0x081F);
  x |= __builtin_amdgcn_ds_swizzle(x, 0x101F);
  x |= __builtin_amdgcn_ds_swizzle(x, 0x201F);
  x |= __builtin_amdgcn_ds_swizzle(x, 0x401F);
  return x;
}

// ---------------------------------------------------------------------------
// 1) fp32 weights -> bf16 workspace. wq pre-scaled by 1/sqrt(HD) = 0.125.
__global__ __launch_bounds__(256) void hfa_convert_weights(
    const float* __restrict__ wq, const float* __restrict__ wk,
    const float* __restrict__ wv, const float* __restrict__ wo,
    __bf16* __restrict__ out) {
  int idx = blockIdx.x * 256 + threadIdx.x;   // 0 .. 4*65536-1
  int sel = idx >> 16, j = idx & 0xFFFF;
  float v = (sel == 0) ? wq[j] * 0.125f
          : (sel == 1) ? wk[j]
          : (sel == 2) ? wv[j] : wo[j];
  out[idx] = (__bf16)v;
}

// ---------------------------------------------------------------------------
// 2) Fused QKV projection: q/k/v = x @ W^T + b, ONE pass over x.
//    One wave = 16 rows x 64 cols x {q,k,v} = 12 WMMA accumulators.
//    q,k stored [N,H,F,HD]; v stored transposed [N,H,HD,F] (for PV B-frags).
__global__ __launch_bounds__(256) void hfa_qkv_proj(
    const float* __restrict__ x, const __bf16* __restrict__ wbf,
    const float* __restrict__ bq, const float* __restrict__ bk,
    const float* __restrict__ bv, __bf16* __restrict__ q,
    __bf16* __restrict__ k, __bf16* __restrict__ vT) {
  const int wave = threadIdx.x >> 5, lane = threadIdx.x & 31;
  const int n16  = lane & 15, hf = lane >> 4;
  const int wid  = blockIdx.x * 8 + wave;      // 0..32767
  const int m0   = (wid >> 2) * 16;            // row tile of M = N*F = 131072
  const int c0   = (wid & 3) * 64;             // col tile of D = 256

  v8f acc[3][4] = {};
  for (int kb = 0; kb < DM; kb += 32) {
    v16bf a;  // A 16x32: lane m=n16; VGPR i holds K = hf*16 + 2i (+1)
    #pragma unroll
    for (int i = 0; i < 8; ++i) {
      int kk = (hf << 4) + 2 * i;
      float2 f2 = *(const float2*)(x + (size_t)(m0 + n16) * DM + kb + kk);
      a[2 * i] = (__bf16)f2.x; a[2 * i + 1] = (__bf16)f2.y;
    }
    #pragma unroll
    for (int sel = 0; sel < 3; ++sel) {
      const __bf16* w = wbf + (size_t)sel * DM * DM;
      #pragma unroll
      for (int ns = 0; ns < 4; ++ns) {
        v16bf b;  // B 32x16: lane n=n16; VGPR i holds K = hf*16 + 2i (+1)
        #pragma unroll
        for (int i = 0; i < 8; ++i) {
          BfPair p;
          p.u = *(const unsigned int*)(w + (size_t)(c0 + ns * 16 + n16) * DM +
                                       kb + (hf << 4) + 2 * i);
          b[2 * i] = p.h[0]; b[2 * i + 1] = p.h[1];
        }
        acc[sel][ns] = wmma_bf16(a, b, acc[sel][ns]);
      }
    }
  }
  // epilogue: bias + store in head-major layouts
  #pragma unroll
  for (int sel = 0; sel < 3; ++sel) {
    const float* bias = (sel == 0) ? bq : (sel == 1) ? bk : bv;
    const float  bsc  = (sel == 0) ? 0.125f : 1.0f;
    #pragma unroll
    for (int ns = 0; ns < 4; ++ns) {
      const int   col = c0 + ns * 16 + n16;
      const float bb  = bias[col] * bsc;
      const int   h = col >> 6, hd = col & 63;
      #pragma unroll
      for (int v = 0; v < 8; ++v) {        // C: element (m = v + 8*hf, n = n16)
        const int row = m0 + v + (hf << 3);
        const int nsq = row >> 9, f = row & 511;
        const float val = acc[sel][ns][v] + bb;
        if (sel == 0)
          q[((size_t)(nsq * NH_ + h) * FB + f) * HD_ + hd] = (__bf16)val;
        else if (sel == 1)
          k[((size_t)(nsq * NH_ + h) * FB + f) * HD_ + hd] = (__bf16)val;
        else
          vT[((size_t)(nsq * NH_ + h) * HD_ + hd) * FB + f] = (__bf16)val;
      }
    }
  }
}

// ---------------------------------------------------------------------------
// 3) Attention. Block = one (n,h); 8 waves = 8 consecutive 16-row q tiles.
//    K [F,HD] and V^T [HD,F] for this (n,h) are staged into LDS once per
//    block via global_load_async_to_lds_b128 (ASYNCcnt DMA), then all score
//    and PV B-fragments are built from LDS.
__global__ __launch_bounds__(256) void hfa_attn(
    const __bf16* __restrict__ q, const __bf16* __restrict__ k,
    const __bf16* __restrict__ vT, const unsigned char* __restrict__ mask,
    __bf16* __restrict__ ob) {
  extern __shared__ __bf16 smem[];
  __bf16* sK = smem;                 // [FB][HD] row-major, 64 KB
  __bf16* sV = smem + FB * HD_;      // [HD][FB], 64 KB
  __bf16* sP = sV + FB * HD_;        // 8 waves x 16x32 P staging, 8 KB

  const int wave = threadIdx.x >> 5, lane = threadIdx.x & 31;
  const int n16  = lane & 15, hf = lane >> 4;
  const int nh   = blockIdx.x >> 2;            // n*H + h in 0..1023
  const int qt   = (blockIdx.x & 3) * 8 + wave;
  const int qb   = qt * 16;
  const __bf16* qp = q  + (size_t)nh * FB * HD_;
  const __bf16* kp = k  + (size_t)nh * FB * HD_;
  const __bf16* vp = vT + (size_t)nh * HD_ * FB;
  __bf16* lp = sP + wave * (16 * 32);

  // --- async DMA: K (64KB) + V (64KB) global -> LDS, 16B per lane per op ---
  {
    const unsigned ldsK = (unsigned)(unsigned long long)(void*)sK;
    const unsigned ldsV = (unsigned)(unsigned long long)(void*)sV;
    const unsigned t = threadIdx.x * 16;       // 256 thr * 16B = 4 KB / iter
    #pragma unroll
    for (int it = 0; it < 16; ++it) {
      const unsigned off = t + it * 4096;
      unsigned long long gk = (unsigned long long)(const char*)kp + off;
      unsigned long long gv = (unsigned long long)(const char*)vp + off;
      unsigned lk = ldsK + off, lv = ldsV + off;
      asm volatile("global_load_async_to_lds_b128 %0, %1, off"
                   :: "v"(lk), "v"(gk) : "memory");
      asm volatile("global_load_async_to_lds_b128 %0, %1, off"
                   :: "v"(lv), "v"(gv) : "memory");
    }
    asm volatile("s_wait_asynccnt 0x0" ::: "memory");
  }
  __syncthreads();

  v8f o[4] = {};
  float mrow[8], lrow[8];
  #pragma unroll
  for (int v = 0; v < 8; ++v) { mrow[v] = -1.0e30f; lrow[v] = 0.0f; }

  for (int kb = 0; kb < FB; kb += 32) {
    // mask block, fetched in score (C) layout: (m = v + 8*hf, n = n16)
    unsigned char mk[16]; int anyb = 0;
    #pragma unroll
    for (int v = 0; v < 8; ++v) {
      const int r = qb + v + (hf << 3);
      mk[2 * v]     = mask[(size_t)r * FB + kb + n16];
      mk[2 * v + 1] = mask[(size_t)r * FB + kb + 16 + n16];
      anyb |= mk[2 * v] | mk[2 * v + 1];
    }
    if (wave_or32(anyb) == 0) continue;        // harmonic band: skip empty tile

    // scores S(16x32) = Q(16x64) @ K_tile^T, K frags from LDS
    v8f s0 = {}, s1 = {};
    #pragma unroll
    for (int st = 0; st < 2; ++st) {
      v16bf aq;
      #pragma unroll
      for (int i = 0; i < 8; ++i) {
        int kk = (hf << 4) + 2 * i;
        BfPair p;
        p.u = *(const unsigned int*)(qp + (size_t)(qb + n16) * HD_ + st * 32 + kk);
        aq[2 * i] = p.h[0]; aq[2 * i + 1] = p.h[1];
      }
      v16bf b0, b1;
      #pragma unroll
      for (int i = 0; i < 8; ++i) {
        int d = st * 32 + (hf << 4) + 2 * i;
        BfPair p0, p1;
        p0.u = *(const unsigned int*)(sK + (kb + n16) * HD_ + d);
        p1.u = *(const unsigned int*)(sK + (kb + 16 + n16) * HD_ + d);
        b0[2 * i] = p0.h[0]; b0[2 * i + 1] = p0.h[1];
        b1[2 * i] = p1.h[0]; b1[2 * i + 1] = p1.h[1];
      }
      s0 = wmma_bf16(aq, b0, s0);
      s1 = wmma_bf16(aq, b1, s1);
    }

    // masked online softmax; write P (bf16) into per-wave LDS slice
    float alpha[8];
    #pragma unroll
    for (int v = 0; v < 8; ++v) {
      float a0 = mk[2 * v]     ? s0[v] : NEGC;
      float a1 = mk[2 * v + 1] ? s1[v] : NEGC;
      float mx   = rowmax16(fmaxf(a0, a1));
      float mnew = fmaxf(mrow[v], mx);
      alpha[v] = __expf(mrow[v] - mnew);
      mrow[v]  = mnew;
      float p0 = __expf(a0 - mnew);
      float p1 = __expf(a1 - mnew);
      lrow[v] = lrow[v] * alpha[v] + rowsum16(p0 + p1);
      const int r = (v + (hf << 3)) * 32;
      lp[r + n16]      = (__bf16)p0;
      lp[r + 16 + n16] = (__bf16)p1;
    }
    #pragma unroll
    for (int ns = 0; ns < 4; ++ns)
      #pragma unroll
      for (int v = 0; v < 8; ++v) o[ns][v] *= alpha[v];

    asm volatile("s_wait_dscnt 0x0" ::: "memory");  // P stores -> A-frag reads

    v16bf pa;                                  // P as A 16x32
    #pragma unroll
    for (int i = 0; i < 8; ++i) {
      int kk = (hf << 4) + 2 * i;
      pa[2 * i]     = lp[n16 * 32 + kk];
      pa[2 * i + 1] = lp[n16 * 32 + kk + 1];
    }
    #pragma unroll
    for (int ns = 0; ns < 4; ++ns) {
      v16bf bvv;                               // V_tile as B 32x16 from LDS sV[HD][F]
      #pragma unroll
      for (int i = 0; i < 8; ++i) {
        BfPair p;
        p.u = *(const unsigned int*)(sV + (ns * 16 + n16) * FB + kb +
                                     (hf << 4) + 2 * i);
        bvv[2 * i] = p.h[0]; bvv[2 * i + 1] = p.h[1];
      }
      o[ns] = wmma_bf16(pa, bvv, o[ns]);
    }
  }

  // normalize and store to ob [N, F, D] (bf16)
  const int nsq = nh >> 2, h = nh & 3;
  #pragma unroll
  for (int v = 0; v < 8; ++v) {
    const float rinv = 1.0f / lrow[v];         // diagonal always unmasked
    const int f = qb + v + (hf << 3);
    #pragma unroll
    for (int ns = 0; ns < 4; ++ns)
      ob[(size_t)(nsq * FB + f) * DM + h * HD_ + ns * 16 + n16] =
          (__bf16)(o[ns][v] * rinv);
  }
}

// ---------------------------------------------------------------------------
// 4) Output projection: out(f32) = ob(bf16) @ Wo^T + bo.
__global__ __launch_bounds__(256) void hfa_out_proj(
    const __bf16* __restrict__ ob, const __bf16* __restrict__ wo,
    const float* __restrict__ bo, float* __restrict__ out) {
  const int wave = threadIdx.x >> 5, lane = threadIdx.x & 31;
  const int n16  = lane & 15, hf = lane >> 4;
  const int wid  = blockIdx.x * 8 + wave;
  const int m0   = (wid >> 2) * 16;
  const int c0   = (wid & 3) * 64;

  v8f acc[4] = {};
  for (int kb = 0; kb < DM; kb += 32) {
    v16bf a;
    #pragma unroll
    for (int i = 0; i < 8; ++i) {
      int kk = (hf << 4) + 2 * i;
      BfPair p;
      p.u = *(const unsigned int*)(ob + (size_t)(m0 + n16) * DM + kb + kk);
      a[2 * i] = p.h[0]; a[2 * i + 1] = p.h[1];
    }
    #pragma unroll
    for (int ns = 0; ns < 4; ++ns) {
      v16bf b;
      #pragma unroll
      for (int i = 0; i < 8; ++i) {
        BfPair p;
        p.u = *(const unsigned int*)(wo + (size_t)(c0 + ns * 16 + n16) * DM +
                                     kb + (hf << 4) + 2 * i);
        b[2 * i] = p.h[0]; b[2 * i + 1] = p.h[1];
      }
      acc[ns] = wmma_bf16(a, b, acc[ns]);
    }
  }
  #pragma unroll
  for (int ns = 0; ns < 4; ++ns) {
    const int   col = c0 + ns * 16 + n16;
    const float bb  = bo[col];
    #pragma unroll
    for (int v = 0; v < 8; ++v) {
      const int row = m0 + v + (hf << 3);
      out[(size_t)row * DM + col] = acc[ns][v] + bb;
    }
  }
}

// ---------------------------------------------------------------------------
extern "C" void kernel_launch(void* const* d_in, const int* in_sizes, int n_in,
                              void* d_out, int out_size, void* d_ws, size_t ws_size,
                              hipStream_t stream) {
  const float* x  = (const float*)d_in[0];
  const float* wq = (const float*)d_in[1];
  const float* wk = (const float*)d_in[2];
  const float* wv = (const float*)d_in[3];
  const float* bq = (const float*)d_in[4];
  const float* bk = (const float*)d_in[5];
  const float* bv = (const float*)d_in[6];
  const float* wo = (const float*)d_in[7];
  const float* bo = (const float*)d_in[8];
  const unsigned char* mask = (const unsigned char*)d_in[9];  // bool8 [F,F]
  float* out = (float*)d_out;

  // Workspace (bf16): [4 weight mats | q | k | vT | ob] = ~257 MB
  __bf16* ws   = (__bf16*)d_ws;
  __bf16* wAll = ws;
  size_t  nqkv = (size_t)NSEQ * NH_ * FB * HD_;   // 33.5M elems each
  __bf16* qb_  = wAll + (size_t)4 * DM * DM;
  __bf16* kb_  = qb_ + nqkv;
  __bf16* vT_  = kb_ + nqkv;
  __bf16* ob_  = vT_ + nqkv;

  // K(64KB) + V(64KB) + P staging(8KB) dynamic LDS for attention
  const size_t attn_lds = (size_t)(FB * HD_ * 2 + 8 * 16 * 32) * sizeof(__bf16);

  hfa_convert_weights<<<1024, 256, 0, stream>>>(wq, wk, wv, wo, wAll);
  hfa_qkv_proj<<<4096, 256, 0, stream>>>(x, wAll, bq, bk, bv, qb_, kb_, vT_);
  hfa_attn<<<4096, 256, attn_lds, stream>>>(qb_, kb_, vT_, mask, ob_);
  hfa_out_proj<<<4096, 256, 0, stream>>>(ob_, wAll + (size_t)3 * DM * DM,
                                         bo, out);
}
